// AugmentedLayer_3315714753199
// MI455X (gfx1250) — compile-verified, matrix operand order checked by reading, not run on hardware
//
#include <hip/hip_runtime.h>
#include <cmath>
#include <cstddef>

// ---------------------------------------------------------------------------
// Types for CDNA5 WMMA (wave32, 16x16x32 bf16 -> f32)
// ---------------------------------------------------------------------------
typedef __attribute__((ext_vector_type(16))) __bf16       v16bf;
typedef __attribute__((ext_vector_type(8)))  float        v8f;
typedef __attribute__((ext_vector_type(8)))  unsigned int v8u;

__device__ __forceinline__ unsigned short f2bf(float f) {
  // round-to-nearest-even float -> bf16
  unsigned u = __float_as_uint(f);
  u += 0x7FFFu + ((u >> 16) & 1u);
  return (unsigned short)(u >> 16);
}

// ---------------------------------------------------------------------------
// GEMM: M[N,64] = X[N,64] @ W[64,64]  (bf16 WMMA, f32 accumulate)
// One 128-thread block (4 waves) computes 64 rows. Each wave owns a 16-row
// stripe and does 4 N-tiles x 2 K-steps of v_wmma_f32_16x16x32_bf16.
// Fragment layouts follow CDNA5 ISA 7.12.2 (16-bit A: lanes 0-15 K 0-7,16-23,
// lanes 16-31 K 8-15,24-31; pairs packed per VGPR). W is stored transposed in
// LDS so each lane's (k,k+1) bf16 pair is one aligned 32-bit LDS read
// (the compiler fuses the 8 pair-reads per fragment into 2x ds_load_b128).
// Epilogue stages the f32 C tile through the same 16KB of LDS and writes
// back with cooperative global_store_b128 (no per-element bounds branches).
// ---------------------------------------------------------------------------
__global__ void __launch_bounds__(128)
gemm64_bf16_wmma(const float* __restrict__ X, const float* __restrict__ W,
                 float* __restrict__ M, int N)
{
  // 16 KB shared: first used as bf16 A tile (8KB) + transposed bf16 W (8KB),
  // then (after a barrier) reused as the f32 C tile [64][64].
  __shared__ __align__(16) unsigned char smem[16384];
  unsigned short* sA = (unsigned short*)smem;          // [64][64] bf16
  unsigned short* sB = (unsigned short*)smem + 4096;   // W^T [n][k] bf16
  float*          sC = (float*)smem;                   // [64][64] f32

  const int tid  = threadIdx.x;
  const int row0 = blockIdx.x * 64;
  const bool full = (row0 + 64 <= N);

  for (int i = tid; i < 4096; i += 128) {
    int k = i >> 6, n = i & 63;
    sB[(n << 6) | k] = f2bf(W[(k << 6) | n]);
  }
  if (full) {
    for (int i = tid; i < 1024; i += 128) {
      // 4 consecutive cols per thread-step: coalesced b128 loads
      const float4 xv = ((const float4*)(X + (size_t)row0 * 64))[i];
      sA[i * 4 + 0] = f2bf(xv.x); sA[i * 4 + 1] = f2bf(xv.y);
      sA[i * 4 + 2] = f2bf(xv.z); sA[i * 4 + 3] = f2bf(xv.w);
    }
  } else {
    for (int i = tid; i < 4096; i += 128) {
      int r = i >> 6, c = i & 63;
      int gr = row0 + r;
      sA[i] = f2bf(gr < N ? X[(size_t)gr * 64 + c] : 0.0f);
    }
  }
  __syncthreads();

  const int wave = tid >> 5, lane = tid & 31;
  const int mloc = lane & 15, kh = lane >> 4;   // kh: K half select per ISA layout
  const int arow = wave * 16 + mloc;

  const v8f vzero = {0.f, 0.f, 0.f, 0.f, 0.f, 0.f, 0.f, 0.f};
  v8f acc[4] = {vzero, vzero, vzero, vzero};

#pragma unroll
  for (int ks = 0; ks < 2; ++ks) {
    const int kbase = ks * 32 + kh * 8;
    v8u au;
#pragma unroll
    for (int v = 0; v < 8; ++v) {
      const int k = kbase + ((v < 4) ? (v * 2) : (16 + (v - 4) * 2));
      au[v] = *(const unsigned int*)&sA[(arow << 6) | k];
    }
    const v16bf a = __builtin_bit_cast(v16bf, au);
#pragma unroll
    for (int nt = 0; nt < 4; ++nt) {
      v8u bu;
      const int n = nt * 16 + mloc;
#pragma unroll
      for (int v = 0; v < 8; ++v) {
        const int k = kbase + ((v < 4) ? (v * 2) : (16 + (v - 4) * 2));
        bu[v] = *(const unsigned int*)&sB[(n << 6) | k];
      }
      const v16bf b = __builtin_bit_cast(v16bf, bu);
      acc[nt] = __builtin_amdgcn_wmma_f32_16x16x32_bf16(
          false, a, false, b, (short)0, acc[nt], false, false);
    }
  }

  // Stage C through LDS (all A/B reads are done; barrier then reuse smem).
  // C/D layout: VGPR r holds row (r + 8*kh), column = nt*16 + (lane&15).
  __syncthreads();
#pragma unroll
  for (int nt = 0; nt < 4; ++nt) {
    const int col = nt * 16 + mloc;
#pragma unroll
    for (int r = 0; r < 8; ++r) {
      sC[((wave * 16 + kh * 8 + r) << 6) | col] = acc[nt][r];
    }
  }
  __syncthreads();

  // Cooperative coalesced write-back: 1024 float4 chunks / 128 threads.
  float* __restrict__ Mblk = M + (size_t)row0 * 64;
  if (full) {
#pragma unroll
    for (int i = tid; i < 1024; i += 128) {
      ((float4*)Mblk)[i] = ((const float4*)sC)[i];
    }
  } else {
    for (int i = tid; i < 1024; i += 128) {
      const int row = i >> 4;
      if (row0 + row < N) ((float4*)Mblk)[i] = ((const float4*)sC)[i];
    }
  }
}

// ---------------------------------------------------------------------------
// Per-node scalars: s1[i] = M[i,:].v1 ; s2[i] = M[i,:].v2  (one wave per node)
// ---------------------------------------------------------------------------
__global__ void node_scalars_kernel(const float* __restrict__ M,
                                    const float* __restrict__ v1,
                                    const float* __restrict__ v2,
                                    float* __restrict__ s1,
                                    float* __restrict__ s2, int N)
{
  const int wid  = (int)((blockIdx.x * (size_t)blockDim.x + threadIdx.x) >> 5);
  const int lane = threadIdx.x & 31;
  if (wid >= N) return;
  const float2 mm = ((const float2*)(M + (size_t)wid * 64))[lane];
  float d1 = 0.f, d2 = 0.f;
  if (v1) { const float2 a = ((const float2*)v1)[lane]; d1 = mm.x * a.x + mm.y * a.y; }
  if (v2) { const float2 a = ((const float2*)v2)[lane]; d2 = mm.x * a.x + mm.y * a.y; }
#pragma unroll
  for (int off = 16; off; off >>= 1) {
    d1 += __shfl_xor(d1, off, 32);
    d2 += __shfl_xor(d2, off, 32);
  }
  if (lane == 0) {
    if (v1) s1[wid] = d1;
    if (v2) s2[wid] = d2;
  }
}

// u[k] = sum_c W[k,c] * v[c]   (collapses xs@Ws@a2 into xs@(Ws@a2))
__global__ void matvec64_kernel(const float* __restrict__ W,
                                const float* __restrict__ v,
                                float* __restrict__ u)
{
  const int k = threadIdx.x;
  float s = 0.f;
#pragma unroll 8
  for (int c = 0; c < 64; ++c) s += W[(k << 6) | c] * v[c];
  u[k] = s;
}

__global__ void fill_kernel(float* __restrict__ p, float val, size_t n)
{
  size_t i = blockIdx.x * (size_t)blockDim.x + threadIdx.x;
  if (i < n) p[i] = val;
}

__global__ void fill4_kernel(float4* __restrict__ p, float val, size_t n4)
{
  size_t i = blockIdx.x * (size_t)blockDim.x + threadIdx.x;
  if (i < n4) p[i] = make_float4(val, val, val, val);
}

// ---------------------------------------------------------------------------
// Edge passes. Generic form covers both HBS and HBNS e/f branches:
//   e_k  = lrelu(sSeg[seg[k]] + sGath[gth[k]])
//   att  = softmax over seg
//   Y[seg[k],:] += att_k * Mg[gth[k],:]
// ---------------------------------------------------------------------------
__global__ void edge_logits_max_kernel(const int* __restrict__ seg,
                                       const int* __restrict__ gth,
                                       const float* __restrict__ sSeg,
                                       const float* __restrict__ sGath,
                                       float* __restrict__ eL,
                                       float* __restrict__ rmax, int E)
{
  const int k = (int)(blockIdx.x * (size_t)blockDim.x + threadIdx.x);
  if (k >= E) return;
  const int s = seg[k];
  float e = sSeg[s] + sGath[gth[k]];
  e = (e > 0.f) ? e : 0.2f * e;           // leaky_relu, slope 0.2
  eL[k] = e;
  // float atomic max via sign-aware integer punning (race-safe for mixed signs)
  if (e >= 0.f) atomicMax((int*)&rmax[s], __float_as_int(e));
  else          atomicMin((unsigned int*)&rmax[s], __float_as_uint(e));
}

__global__ void edge_expsum_kernel(const int* __restrict__ seg,
                                   const float* __restrict__ rmax,
                                   float* __restrict__ eL,
                                   float* __restrict__ rsum, int E)
{
  const int k = (int)(blockIdx.x * (size_t)blockDim.x + threadIdx.x);
  if (k >= E) return;
  const int s = seg[k];
  const float ex = __expf(eL[k] - rmax[s]);
  eL[k] = ex;
  atomicAdd(&rsum[s], ex);                 // global_atomic_add_f32
}

// One wave per edge: lanes cover 64 cols as float2. 256B coalesced gather,
// contiguous f32 atomic adds into the segment row.
__global__ void edge_aggregate_kernel(const int* __restrict__ seg,
                                      const int* __restrict__ gth,
                                      const float* __restrict__ eL,
                                      const float* __restrict__ rsum,
                                      const float* __restrict__ Mg,
                                      float* __restrict__ Y, int E)
{
  const int w    = (int)((blockIdx.x * (size_t)blockDim.x + threadIdx.x) >> 5);
  const int lane = threadIdx.x & 31;
  if (w >= E) return;
  const int s = seg[w], g = gth[w];
  const float att = eL[w] / rsum[s];
  const float2 mv = ((const float2*)(Mg + (size_t)g * 64))[lane];
  float* yrow = Y + (size_t)s * 64 + 2 * lane;
  atomicAdd(yrow,     att * mv.x);
  atomicAdd(yrow + 1, att * mv.y);
}

// ---------------------------------------------------------------------------
// Host-side orchestration (all launches on `stream`; graph-capture safe).
// ---------------------------------------------------------------------------
namespace {

struct Bump {
  char*  base;
  size_t off;
  float* falloc(size_t nfloats) {
    size_t bytes = ((nfloats * sizeof(float)) + 255) & ~(size_t)255;
    float* p = (float*)(base + off);
    off += bytes;
    return p;
  }
};

inline void run_fill(float* p, float v, size_t n, hipStream_t s) {
  if ((n & 3) == 0) {
    const size_t n4 = n >> 2;
    fill4_kernel<<<dim3((unsigned)((n4 + 255) / 256)), 256, 0, s>>>((float4*)p, v, n4);
  } else {
    fill_kernel<<<dim3((unsigned)((n + 255) / 256)), 256, 0, s>>>(p, v, n);
  }
}

inline void run_gemm(const float* X, const float* W, float* M, int N, hipStream_t s) {
  gemm64_bf16_wmma<<<dim3((unsigned)((N + 63) / 64)), 128, 0, s>>>(X, W, M, N);
}

inline void run_scalars(const float* M, const float* v1, const float* v2,
                        float* s1, float* s2, int N, hipStream_t s) {
  node_scalars_kernel<<<dim3((unsigned)((N + 7) / 8)), 256, 0, s>>>(M, v1, v2, s1, s2, N);
}

// softmax-over-seg attention aggregation; Y must be pre-zeroed by caller.
inline void run_attention(const int* seg, const int* gth,
                          const float* sSeg, const float* sGath,
                          const float* Mg, float* Y, int nseg, int E,
                          Bump& ar, hipStream_t s) {
  float* eL   = ar.falloc((size_t)E);
  float* rmax = ar.falloc((size_t)nseg);
  float* rsum = ar.falloc((size_t)nseg);
  run_fill(rmax, -INFINITY, (size_t)nseg, s);
  run_fill(rsum, 0.0f,      (size_t)nseg, s);
  const unsigned gE = (unsigned)((E + 255) / 256);
  edge_logits_max_kernel<<<gE, 256, 0, s>>>(seg, gth, sSeg, sGath, eL, rmax, E);
  edge_expsum_kernel<<<gE, 256, 0, s>>>(seg, rmax, eL, rsum, E);
  edge_aggregate_kernel<<<dim3((unsigned)((E + 7) / 8)), 256, 0, s>>>(seg, gth, eL, rsum, Mg, Y, E);
}

// HBS: Y[row] += softmax_row(lrelu(s1[row]+s2[col])) * (X@W)[col]
inline void run_hbs(const float* X, const float* W, const float* a,
                    const int* row, const int* col, float* Y,
                    int N, int E, Bump ar, hipStream_t s) {
  float* m  = ar.falloc((size_t)N * 64);
  float* s1 = ar.falloc((size_t)N);
  float* s2 = ar.falloc((size_t)N);
  run_gemm(X, W, m, N, s);
  run_scalars(m, a, a + 64, s1, s2, N, s);
  run_attention(row, col, s1, s2, m, Y, N, E, ar, s);
}

// HBNS, both branches (layer 1): accumulates msg_t into Yt, msg_s into Ys.
inline void run_hbns_full(const float* xs, const float* xt,
                          const int* ti, const int* sj,
                          const float* Ws, const float* Wt, const float* a,
                          float* Yt, float* Ys, int ns, int nt, int E,
                          Bump ar, hipStream_t s) {
  float* sm  = ar.falloc((size_t)ns * 64);
  float* tm  = ar.falloc((size_t)nt * 64);
  float* smA = ar.falloc((size_t)ns);
  float* smB = ar.falloc((size_t)ns);
  float* tmA = ar.falloc((size_t)nt);
  float* tmB = ar.falloc((size_t)nt);
  run_gemm(xs, Ws, sm, ns, s);
  run_gemm(xt, Wt, tm, nt, s);
  run_scalars(sm, a, a + 64, smA, smB, ns, s);
  run_scalars(tm, a, a + 64, tmA, tmB, nt, s);
  // e-branch: seg=ti (nt side), e=lrelu(smA[sj]+tmB[ti]); msg_t += att*sm[sj]
  run_attention(ti, sj, tmB, smA, sm, Yt, nt, E, ar, s);
  // f-branch: seg=sj (ns side), f=lrelu(tmA[ti]+smB[sj]); msg_s += att*tm[ti]
  run_attention(sj, ti, smB, tmA, tm, Ys, ns, E, ar, s);
}

// HBNS, msg_s only (layer 2). sm is only needed through sm@a2, so fold the
// full [N,64]x[64,64] GEMM into a 64-vector matvec: smB = xs @ (Ws@a2).
inline void run_hbns_s(const float* xs, const float* xt,
                       const int* ti, const int* sj,
                       const float* Ws, const float* Wt, const float* a,
                       float* Ys, int ns, int nt, int E,
                       Bump ar, hipStream_t s) {
  float* tm  = ar.falloc((size_t)nt * 64);
  float* tmA = ar.falloc((size_t)nt);
  float* smB = ar.falloc((size_t)ns);
  float* u   = ar.falloc(64);
  run_gemm(xt, Wt, tm, nt, s);
  run_scalars(tm, a, nullptr, tmA, nullptr, nt, s);
  matvec64_kernel<<<1, 64, 0, s>>>(Ws, a + 64, u);
  run_scalars(xs, u, nullptr, smB, nullptr, ns, s);
  run_attention(sj, ti, smB, tmA, tm, Ys, ns, E, ar, s);
}

}  // namespace

extern "C" void kernel_launch(void* const* d_in, const int* in_sizes, int n_in,
                              void* d_out, int out_size, void* d_ws, size_t ws_size,
                              hipStream_t stream) {
  const float* x0   = (const float*)d_in[0];
  const float* x1   = (const float*)d_in[1];
  const float* x2   = (const float*)d_in[2];
  const float* x3   = (const float*)d_in[3];
  const float* hbsW = (const float*)d_in[4];   // [5][64][64]
  const float* hbsA = (const float*)d_in[5];   // [5][128]
  const float* hWs  = (const float*)d_in[6];   // [6][64][64]
  const float* hWt  = (const float*)d_in[7];   // [6][64][64]
  const float* hA   = (const float*)d_in[8];   // [6][128]
  const int* adj0r = (const int*)d_in[9],  *adj0c = (const int*)d_in[10];
  const int* adj1r = (const int*)d_in[11], *adj1c = (const int*)d_in[12];
  const int* adj2r = (const int*)d_in[13], *adj2c = (const int*)d_in[14];
  const int* co3r  = (const int*)d_in[15], *co3c  = (const int*)d_in[16];
  const int* i1r   = (const int*)d_in[17], *i1c   = (const int*)d_in[18];
  const int* i2r   = (const int*)d_in[19], *i2c   = (const int*)d_in[20];
  const int* i3r   = (const int*)d_in[21], *i3c   = (const int*)d_in[22];

  const int N0 = in_sizes[0] / 64, N1 = in_sizes[1] / 64;
  const int N2 = in_sizes[2] / 64, N3 = in_sizes[3] / 64;
  const int E_a0 = in_sizes[9],  E_a1 = in_sizes[11], E_a2 = in_sizes[13];
  const int E_c3 = in_sizes[15], E_i1 = in_sizes[17], E_i2 = in_sizes[19];
  const int E_i3 = in_sizes[21];

  // Persistent region: layer-1 accumulators (contiguous, single zero fill).
  Bump persist{(char*)d_ws, 0};
  const size_t nAll = (size_t)(N0 + N1 + N2 + N3) * 64;
  float* XL1  = persist.falloc(nAll);
  float* X0L1 = XL1;
  float* X1L1 = X0L1 + (size_t)N0 * 64;
  float* X2L1 = X1L1 + (size_t)N1 * 64;
  float* X3L1 = X2L1 + (size_t)N2 * 64;
  // Transient arena: reset (by pass-by-value Bump) per block; stream ordering
  // makes sequential reuse safe.
  Bump scratch{(char*)d_ws + persist.off, 0};

  run_fill(XL1, 0.0f, nAll, stream);

  // ---- Layer 1 ----
  // x00 -> X0L1 ; hbns1: msg_t(x10)->X0L1, msg_s(x01)->X1L1 ; etc.
  run_hbs(x0, hbsW + 0 * 4096, hbsA + 0 * 128, adj0r, adj0c, X0L1, N0, E_a0, scratch, stream);
  run_hbns_full(x1, x0, i1r, i1c, hWs + 0 * 4096, hWt + 0 * 4096, hA + 0 * 128,
                X0L1, X1L1, N1, N0, E_i1, scratch, stream);
  run_hbns_full(x2, x1, i2r, i2c, hWs + 1 * 4096, hWt + 1 * 4096, hA + 1 * 128,
                X1L1, X2L1, N2, N1, E_i2, scratch, stream);
  run_hbns_full(x3, x2, i3r, i3c, hWs + 2 * 4096, hWt + 2 * 4096, hA + 2 * 128,
                X2L1, X3L1, N3, N2, E_i3, scratch, stream);

  // ---- Layer 2 (outputs accumulate directly into d_out regions) ----
  float* O0 = (float*)d_out;
  float* O1 = O0 + (size_t)N0 * 64;
  float* O2 = O1 + (size_t)N1 * 64;
  float* O3 = O2 + (size_t)N2 * 64;
  run_fill(O0, 0.0f, nAll, stream);

  run_hbs(X0L1, hbsW + 1 * 4096, hbsA + 1 * 128, adj0r, adj0c, O0, N0, E_a0, scratch, stream);
  run_hbs(X1L1, hbsW + 2 * 4096, hbsA + 2 * 128, adj1r, adj1c, O1, N1, E_a1, scratch, stream);
  run_hbns_s(X1L1, X0L1, i1r, i1c, hWs + 3 * 4096, hWt + 3 * 4096, hA + 3 * 128,
             O1, N1, N0, E_i1, scratch, stream);
  run_hbs(X2L1, hbsW + 3 * 4096, hbsA + 3 * 128, adj2r, adj2c, O2, N2, E_a2, scratch, stream);
  run_hbns_s(X2L1, X1L1, i2r, i2c, hWs + 4 * 4096, hWt + 4 * 4096, hA + 4 * 128,
             O2, N2, N1, E_i2, scratch, stream);
  run_hbs(X3L1, hbsW + 4 * 4096, hbsA + 4 * 128, co3r, co3c, O3, N3, E_c3, scratch, stream);
  run_hbns_s(X3L1, X2L1, i3r, i3c, hWs + 5 * 4096, hWt + 5 * 4096, hA + 5 * 128,
             O3, N3, N2, E_i3, scratch, stream);

  (void)n_in; (void)out_size; (void)ws_size;
}
